// DualEncoderUNetPP_61194694034049
// MI455X (gfx1250) — matrix operand order repaired and checked by reading, not compiled, not run on hardware
//
#include <hip/hip_runtime.h>

typedef __attribute__((ext_vector_type(16))) _Float16 v16h;
typedef __attribute__((ext_vector_type(2)))  _Float16 v2h;
typedef __attribute__((ext_vector_type(2)))  __fp16   v2hf;   // cvt_pkrtz result type
typedef __attribute__((ext_vector_type(8)))  float    v8f;
typedef __attribute__((ext_vector_type(4)))  int      v4i;

#define HW      (512*512)
#define NC      16
#define DD      128
#define NTILES  32768          // (4*HW/16)/2 : wave-iterations, 2 pixel-tiles each

// fragment indices in d_ws (each fragment = 32 lanes * 32 bytes = 1024 B, f16 A-layout)
#define FRAG_W0     0          // 8 mt * 4 kt
#define FRAG_W1     32
#define GFRAG_FEAT  64         // 8 mt (K=16 data + feat_b at K=16 slot)
#define GFRAG_MAGG0 72         // 8 mt (MaggT + gnn_b0 at K=16)
#define GFRAG_MAGG1 80
#define GFRAG_OUT   88         // 4 kt
#define NFRAG_TOTAL 92
#define LDS_BYTES   (64*1024)  // W0 + W1 fragments

union F16Frag {
    v16h v;
    v2h  h2[8];
    unsigned u[8];
    v4i  i4[2];
};

__device__ __forceinline__ unsigned pk16(float a, float b) {
    v2hf t = __builtin_amdgcn_cvt_pkrtz(a, b);
    return __builtin_bit_cast(unsigned, t);
}

// lane i <-> lane i^16 swap via v_permlanex16_b32 (identity selects), pure VALU.
__device__ __forceinline__ unsigned xor16u(unsigned x) {
    return __builtin_amdgcn_permlanex16(x, x, 0x76543210u, 0xFEDCBA98u, false, false);
}
__device__ __forceinline__ float xor16f(float x) {
    unsigned u = __builtin_bit_cast(unsigned, x);
    return __builtin_bit_cast(float, xor16u(u));
}

__device__ __forceinline__ v8f wmma16(const F16Frag& a, const F16Frag& b, v8f c) {
    return __builtin_amdgcn_wmma_f32_16x16x32_f16(false, a.v, false, b.v,
                                                  (short)0, c, false, false);
}

__device__ __forceinline__ F16Frag loadFragLDS(const char* sm, int f, int lane) {
    const v4i* p = (const v4i*)(sm + f * 1024 + lane * 32);
    F16Frag r;
    r.i4[0] = p[0];
    r.i4[1] = p[1];
    return r;
}

__device__ __forceinline__ F16Frag loadFragG(const _Float16* fr, int f, int lane) {
    const v4i* p = (const v4i*)((const char*)fr + f * 1024 + lane * 32);
    F16Frag r;
    r.i4[0] = p[0];
    r.i4[1] = p[1];
    return r;
}

// Build B-fragment [K=32, N=16] from 8 per-lane f32 values (classes 8*(lane/16)+i
// of pixel lane%16). Lanes 16..31 carry the K=16..31 pad rows: K=16 row = 1.0
// (bias multiplier), rest zero. B layout: lanes 0-15 hold K=0..15 (2/VGPR),
// lanes 16-31 hold K=16..31.
__device__ __forceinline__ void buildBClass(const float* v, bool lo, F16Frag& f) {
#pragma unroll
    for (int i = 0; i < 4; ++i) {
        unsigned own = pk16(v[2 * i], v[2 * i + 1]);
        unsigned ot  = xor16u(own);
        f.u[i]     = lo ? own : (i == 0 ? 0x00003C00u : 0u);  // half 1.0 at K=16
        f.u[4 + i] = lo ? ot : 0u;
    }
}

// Convert two f32 D accumulator tiles (rows 32k..32k+15 and 32k+16..31) into one
// f16 B fragment for the next layer. D: VGPR r, lane l -> row r+8*(l/16), col l%16.
template <bool RELU>
__device__ __forceinline__ void convD2B(const v8f& dl, const v8f& dh, bool lo, F16Frag& f) {
    float p[8], q[8];
#pragma unroll
    for (int r = 0; r < 8; ++r) {
        float a  = dl[r];
        float b  = dh[r];
        float zs = lo ? b : a;
        float z  = xor16f(zs);
        float pp = lo ? a : z;   // K = r (lanes<16) / 16+r (lanes>=16)
        float qq = lo ? z : b;   // K = 8+r         / 24+r
        if (RELU) { pp = fmaxf(pp, 0.f); qq = fmaxf(qq, 0.f); }
        p[r] = pp;
        q[r] = qq;
    }
#pragma unroll
    for (int v = 0; v < 4; ++v) {
        f.u[v]     = pk16(p[2 * v], p[2 * v + 1]);
        f.u[4 + v] = pk16(q[2 * v], q[2 * v + 1]);
    }
}

// One GNN layer, pair-at-a-time to keep only 4 accumulator tiles live:
// for each output pair (mt=2p,2p+1): acc = [MaggT|b]*[probs;1] + W*Xin, then
// immediately convert the pair into Xout[p].
template <bool RELU>
__device__ __forceinline__ void layerStep(const char* sm, const _Float16* frags,
                                          int baseW, int baseM, bool lo, int lane,
                                          const F16Frag (&Pf)[2],
                                          const F16Frag (&Xin)[2][4],
                                          F16Frag (&Xout)[2][4]) {
#pragma unroll
    for (int p = 0; p < 4; ++p) {
        v8f d[2][2];
#pragma unroll
        for (int s = 0; s < 2; ++s) {
            const int mt = 2 * p + s;
            F16Frag a = loadFragG(frags, baseM + mt, lane);
            v8f z = {};
            d[0][s] = wmma16(a, Pf[0], z);
            d[1][s] = wmma16(a, Pf[1], z);
#pragma unroll
            for (int kt = 0; kt < 4; ++kt) {
                F16Frag w = loadFragLDS(sm, baseW + mt * 4 + kt, lane);
                d[0][s] = wmma16(w, Xin[0][kt], d[0][s]);
                d[1][s] = wmma16(w, Xin[1][kt], d[1][s]);
            }
        }
        convD2B<RELU>(d[0][0], d[0][1], lo, Xout[0][p]);
        convD2B<RELU>(d[1][0], d[1][1], lo, Xout[1][p]);
    }
}

// ---------------------------------------------------------------------------
// Prep kernel: adjacency MLP + Magg = adj @ sem, then write all weight
// fragments pre-swizzled in the WMMA f16 A-layout into d_ws.
// ---------------------------------------------------------------------------
__global__ __launch_bounds__(256) void gnn_prep(
    const float* __restrict__ sem, const float* __restrict__ aw1, const float* __restrict__ ab1,
    const float* __restrict__ aw2, const float* __restrict__ ab2,
    const float* __restrict__ w0, const float* __restrict__ b0,
    const float* __restrict__ w1, const float* __restrict__ b1,
    const float* __restrict__ fw, const float* __restrict__ fb,
    const float* __restrict__ ow, _Float16* __restrict__ frag) {
    __shared__ float e1[16][64], e2[16][64], adj[16][16], magg[16][128], rs[16];
    int tid = threadIdx.x;

    for (int idx = tid; idx < 16 * 64; idx += 256) {
        int r = idx >> 6, c = idx & 63;
        float s1 = ab1[c], s2 = ab2[c];
        for (int k = 0; k < 128; ++k) {
            float sv = sem[r * 128 + k];
            s1 += sv * aw1[c * 128 + k];
            s2 += sv * aw2[c * 128 + k];
        }
        e1[r][c] = fmaxf(s1, 0.f);
        e2[r][c] = fmaxf(s2, 0.f);
    }
    __syncthreads();
    {
        int r = tid >> 4, c = tid & 15;
        float s = 0.f;
        for (int k = 0; k < 64; ++k) s += e1[r][k] * e2[c][k];
        float a = 1.f / (1.f + __expf(-s));
        if (r == c) a += 1.f;
        adj[r][c] = a;
    }
    __syncthreads();
    if (tid < 16) {
        float s = 0.f;
        for (int c = 0; c < 16; ++c) s += adj[tid][c];
        rs[tid] = 1.f / s;
    }
    __syncthreads();
    {
        int r = tid >> 4, c = tid & 15;
        adj[r][c] *= rs[r];
    }
    __syncthreads();
    for (int idx = tid; idx < 16 * 128; idx += 256) {
        int c = idx >> 7, d = idx & 127;
        float s = 0.f;
        for (int k = 0; k < 16; ++k) s += adj[c][k] * sem[k * 128 + d];
        magg[c][d] = s;
    }
    __syncthreads();

    // A-layout as packed here (16-bit, 16x32, contiguous k, B matches): lane<16: K=h
    //                           lane>=16: K=h+16
    for (int idx = tid; idx < NFRAG_TOTAL * 512; idx += 256) {
        int f = idx >> 9;
        int wi = idx & 511;
        int l = wi >> 4;
        int h = wi & 15;
        int m = l & 15;
        int K = (l < 16) ? h : h + 16;
        float v = 0.f;
        if (f < 32) {                       // W0
            int mt = f >> 2, kt = f & 3;
            v = w0[(mt * 16 + m) * 128 + kt * 32 + K];
        } else if (f < 64) {                // W1
            int q = f - 32;
            int mt = q >> 2, kt = q & 3;
            v = w1[(mt * 16 + m) * 128 + kt * 32 + K];
        } else if (f < 72) {                // FEAT (+feat_b at K=16)
            int d0 = (f - 64) * 16 + m;
            if (K < 16) v = fw[d0 * 16 + K];
            else if (K == 16) v = fb[d0];
        } else if (f < 80) {                // MAGG0 (+gnn_b0 at K=16)
            int d0 = (f - 72) * 16 + m;
            if (K < 16) v = magg[K][d0];
            else if (K == 16) v = b0[d0];
        } else if (f < 88) {                // MAGG1 (+gnn_b1 at K=16)
            int d0 = (f - 80) * 16 + m;
            if (K < 16) v = magg[K][d0];
            else if (K == 16) v = b1[d0];
        } else {                            // OUT_W
            int kt = f - 88;
            v = ow[m * 128 + kt * 32 + K];
        }
        frag[idx] = (_Float16)v;
    }
}

// ---------------------------------------------------------------------------
// Main kernel: each wave processes 2 tiles of 16 pixels per iteration.
// ---------------------------------------------------------------------------
__global__ __launch_bounds__(256) void gnn_main(
    const float* __restrict__ logits, const _Float16* __restrict__ frags,
    const float* __restrict__ out_b, const float* __restrict__ gatep,
    float* __restrict__ out) {
    extern __shared__ v4i smv[];
    char* sm = (char*)smv;
    // Stage W0/W1 fragments (64 KB) into LDS
    for (int i = threadIdx.x * 16; i < LDS_BYTES; i += 256 * 16)
        *(v4i*)(sm + i) = *(const v4i*)((const char*)frags + i);
    __syncthreads();

    const int lane  = threadIdx.x & 31;
    const bool lo   = lane < 16;
    const int half8 = (lane >> 4) * 8;      // class group base: 0 or 8
    const int j     = lane & 15;            // pixel within tile
    const float gate = gatep[0];
    float gob[8];
#pragma unroll
    for (int r = 0; r < 8; ++r) gob[r] = gate * out_b[half8 + r];

    const int wgid = blockIdx.x * 8 + (threadIdx.x >> 5);
    const int nw   = gridDim.x * 8;

    for (int it = wgid; it < NTILES; it += nw) {
        const int P0  = it << 5;            // 32 pixels
        const int bb  = P0 >> 18;           // / HW
        const int hw0 = P0 & (HW - 1);
        const float* lb = logits + (size_t)bb * (NC * HW) + hw0;
        float* obase    = out    + (size_t)bb * (NC * HW) + hw0;

        // Load logits: lane holds classes half8..half8+7 of pixel j (per tile t)
        float lg[2][8];
#pragma unroll
        for (int t = 0; t < 2; ++t)
#pragma unroll
            for (int i = 0; i < 8; ++i)
                lg[t][i] = lb[(size_t)(half8 + i) * HW + t * 16 + j];

        // softmax over classes (8 local + pair lane^16), build probs B frag.
        // Max-subtracted arg is in [-inf, 0] and logits are O(1), so the raw
        // v_exp_f32 (__builtin_amdgcn_exp2f) is safe; v_rcp_f32 for the norm.
        F16Frag Pf[2];
#pragma unroll
        for (int t = 0; t < 2; ++t) {
            float pr[8];
            float mx = lg[t][0];
#pragma unroll
            for (int i = 1; i < 8; ++i) mx = fmaxf(mx, lg[t][i]);
            mx = fmaxf(mx, xor16f(mx));
            float s = 0.f;
#pragma unroll
            for (int i = 0; i < 8; ++i) {
                float e = __builtin_amdgcn_exp2f((lg[t][i] - mx) * 1.44269504f);
                pr[i] = e;
                s += e;
            }
            s += xor16f(s);
            float inv = __builtin_amdgcn_rcpf(s);
#pragma unroll
            for (int i = 0; i < 8; ++i) pr[i] *= inv;
            buildBClass(pr, lo, Pf[t]);
        }

        F16Frag Xa[2][4], Xb[2][4];

        // ---- feat projection: X0 = [feat_w | feat_b] * [L ; 1], pair-at-a-time
        {
            F16Frag Lf[2];
#pragma unroll
            for (int t = 0; t < 2; ++t) buildBClass(lg[t], lo, Lf[t]);
#pragma unroll
            for (int p = 0; p < 4; ++p) {
                F16Frag a0 = loadFragG(frags, GFRAG_FEAT + 2 * p, lane);
                F16Frag a1 = loadFragG(frags, GFRAG_FEAT + 2 * p + 1, lane);
                v8f z = {};
                v8f d00 = wmma16(a0, Lf[0], z);
                v8f d01 = wmma16(a1, Lf[0], z);
                v8f d10 = wmma16(a0, Lf[1], z);
                v8f d11 = wmma16(a1, Lf[1], z);
                convD2B<false>(d00, d01, lo, Xa[0][p]);
                convD2B<false>(d10, d11, lo, Xa[1][p]);
            }
        }

        // ---- GNN layer 0 (relu): Xb = relu(W0*Xa + [MaggT|b0]*[probs;1])
        layerStep<true>(sm, frags, FRAG_W0, GFRAG_MAGG0, lo, lane, Pf, Xa, Xb);
        // ---- GNN layer 1: Xa = W1*Xb + [MaggT|b1]*[probs;1]
        layerStep<false>(sm, frags, FRAG_W1, GFRAG_MAGG1, lo, lane, Pf, Xb, Xa);

        // ---- output projection: R = out_w * X2
        v8f rA = {}, rB = {};
#pragma unroll
        for (int kt = 0; kt < 4; ++kt) {
            F16Frag w = loadFragG(frags, GFRAG_OUT + kt, lane);
            rA = wmma16(w, Xa[0][kt], rA);
            rB = wmma16(w, Xa[1][kt], rB);
        }

        // ---- epilogue: out = gate*R + gate*out_b + logits (D layout matches lg)
#pragma unroll
        for (int r = 0; r < 8; ++r) {
            obase[(size_t)(half8 + r) * HW + j]      = fmaf(gate, rA[r], gob[r] + lg[0][r]);
            obase[(size_t)(half8 + r) * HW + 16 + j] = fmaf(gate, rB[r], gob[r] + lg[1][r]);
        }
    }
}

extern "C" void kernel_launch(void* const* d_in, const int* in_sizes, int n_in,
                              void* d_out, int out_size, void* d_ws, size_t ws_size,
                              hipStream_t stream) {
    const float* logits = (const float*)d_in[0];
    const float* sem    = (const float*)d_in[1];
    const float* aw1    = (const float*)d_in[2];
    const float* ab1    = (const float*)d_in[3];
    const float* aw2    = (const float*)d_in[4];
    const float* ab2    = (const float*)d_in[5];
    const float* w0     = (const float*)d_in[6];
    const float* b0     = (const float*)d_in[7];
    const float* w1     = (const float*)d_in[8];
    const float* b1     = (const float*)d_in[9];
    const float* fw     = (const float*)d_in[10];
    const float* fb     = (const float*)d_in[11];
    const float* ow     = (const float*)d_in[12];
    const float* obv    = (const float*)d_in[13];
    const float* gate   = (const float*)d_in[14];
    _Float16* frags     = (_Float16*)d_ws;

    gnn_prep<<<1, 256, 0, stream>>>(sem, aw1, ab1, aw2, ab2, w0, b0, w1, b1,
                                    fw, fb, ow, frags);
    gnn_main<<<2048, 256, LDS_BYTES, stream>>>(logits, frags, obv, gate,
                                               (float*)d_out);
}
